// ContextualAttentionModule_66855460929954
// MI455X (gfx1250) — compile-verified, matrix operand order checked by reading, not run on hardware
//
#include <hip/hip_runtime.h>
#include <hip/hip_bf16.h>

// ---------------------------------------------------------------------------
// Contextual attention module for MI455X (gfx1250, wave32, WMMA).
// Pipeline:
//   prep      : f32 NCHW -> f16 channels-last padded fg/bg (+mask), plus
//               3 x-shifted channel-major copies of bg for aligned B-frags
//   wt        : conv_w [4,64,256,3,3] f32 -> [4][64][2304] f16 rows
//   norm      : per-patch L2 norms of bg patches -> 1/norm
//   gemm1     : scores[b][hw][l] = im2col(fg) . im2col(bg)^T  (WMMA f16)
//   softmax   : 3x3 window-sum over space + softmax over l + fold 1/norm
//   gemm2     : recovered = attn (x) bg-patches (transposed conv as WMMA),
//               fused mask-composite -> padded f16 "final"
//   gemm3     : 4 dilated 3x3 convs as WMMA GEMMs, fused bias+ReLU -> out
// ---------------------------------------------------------------------------

typedef __attribute__((ext_vector_type(16))) _Float16 v16h;
typedef __attribute__((ext_vector_type(8)))  float    v8f;

#define BZ    8
#define CC    256
#define HH    32
#define HW    1024
#define PW    34          // padded width (pad 1)
#define P2    1156        // 34*34
#define FP    48          // pad-8 width for dilated convs
#define FP2   2304        // 48*48
#define KTOT  2304        // 9*256

// workspace offsets (bytes), all 256-aligned
#define OFF_FGCL   ((size_t)0)                       // [BZ][P2][CC] f16  = 4,734,976
#define OFF_BGCL   ((size_t)4734976)                 // [BZ][P2][CC] f16
#define OFF_BGTS   ((size_t)9469952)                 // [3][BZ][CC][34][32] f16 = 13,369,344
#define OFF_NORM   ((size_t)22839296)                // [BZ][1024] f32
#define OFF_SCORE  ((size_t)22872064)                // [BZ][1024][1024] f32 = 33,554,432
#define OFF_ATTNP  ((size_t)56426496)                // [BZ][P2][1024] f16 = 18,939,904
#define OFF_FINAL  ((size_t)75366400)                // [BZ][FP2][CC] f16 = 9,437,184
#define OFF_WT     ((size_t)84803584)                // [4][64][2304] f16 = 1,179,648

static __device__ __forceinline__ v16h ldfrag(const _Float16* p) {
    return *(const v16h*)p;  // 32B, alignment guaranteed by layout
}
static __device__ __forceinline__ v8f wmma_f16(v16h a, v16h b, v8f c) {
    return __builtin_amdgcn_wmma_f32_16x16x32_f16(false, a, false, b,
                                                  (short)0, c, false, false);
}

// ------------------------------- utility -----------------------------------
__global__ void zero_kernel(unsigned int* p, size_t n) {
    size_t i = (size_t)blockIdx.x * blockDim.x + threadIdx.x;
    size_t st = (size_t)gridDim.x * blockDim.x;
    for (; i < n; i += st) p[i] = 0u;
}

// ------------------------------- prep --------------------------------------
// grid (P2, BZ), 256 thr. Builds padded channels-last fg/bg (bg masked) and
// three x-shifted channel-major bg copies for aligned GEMM2 B fragments.
__global__ void prep_kernel(const float* __restrict__ fg,
                            const float* __restrict__ bg,
                            const float* __restrict__ mask,
                            _Float16* __restrict__ fgcl,
                            _Float16* __restrict__ bgcl,
                            _Float16* __restrict__ bgTs) {
    const int b = blockIdx.y, p = blockIdx.x;
    const int y = p / PW, x = p % PW;
    const int c = threadIdx.x;
    float fv = 0.f, bv = 0.f;
    if (y >= 1 && y <= HH && x >= 1 && x <= HH) {
        const int h = y - 1, w = x - 1;
        const size_t idx = (((size_t)b * CC + c) * HH + h) * HH + w;
        const float mk = mask[h * HH + w];
        fv = fg[idx];
        bv = bg[idx] * (1.f - mk);
    }
    const _Float16 fh = (_Float16)fv, bh = (_Float16)bv;
    const size_t o = ((size_t)b * P2 + p) * CC + c;
    fgcl[o] = fh;
    bgcl[o] = bh;
#pragma unroll
    for (int dx = 0; dx < 3; ++dx) {
        const int lx = x - dx;
        if (lx >= 0 && lx < HH)
            bgTs[((((size_t)dx * BZ + b) * CC + c) * PW + y) * HH + lx] = bh;
    }
}

// ------------------------------- weights -----------------------------------
// conv_w [4][64][256][3][3] -> wt[go][s*256+c] f16 rows (go = g*64+oc)
__global__ void wt_kernel(const float* __restrict__ cw, _Float16* __restrict__ wt) {
    const int idx = blockIdx.x * 256 + threadIdx.x;
    if (idx >= 4 * 64 * KTOT) return;
    const int k = idx % KTOT, go = idx / KTOT;
    const int s = k >> 8, c = k & 255;
    wt[idx] = (_Float16)cw[((size_t)go * CC + c) * 9 + s];
}

// ------------------------------- patch norms -------------------------------
// grid (1024, BZ), 256 thr. norm_inv[b][l] = 1/sqrt(sum (patch+1e-7)^2)
__global__ void norm_kernel(const _Float16* __restrict__ bgcl,
                            float* __restrict__ norm_inv) {
    const int b = blockIdx.y, l = blockIdx.x;
    const int ly = l >> 5, lx = l & 31;
    const int c = threadIdx.x;
    const _Float16* base = bgcl + (size_t)b * P2 * CC;
    float s = 0.f;
#pragma unroll
    for (int dy = 0; dy < 3; ++dy)
#pragma unroll
        for (int dx = 0; dx < 3; ++dx) {
            float v = (float)base[(size_t)((ly + dy) * PW + lx + dx) * CC + c] + 1e-7f;
            s += v * v;
        }
#pragma unroll
    for (int o = 16; o > 0; o >>= 1) s += __shfl_xor(s, o, 32);
    __shared__ float red[8];
    if ((threadIdx.x & 31) == 0) red[threadIdx.x >> 5] = s;
    __syncthreads();
    if (threadIdx.x == 0) {
        float t = 0.f;
#pragma unroll
        for (int i = 0; i < 8; ++i) t += red[i];
        norm_inv[b * HW + l] = rsqrtf(t);
    }
}

// ------------------------------- GEMM1 -------------------------------------
// scores[b][m][l] = sum_s sum_c fgcl[row(m,s)][c] * bgcl[row(l,s)][c] * ninv[l]
// block: 256 thr = 8 waves (2M x 4N); block tile 128x128; wave 64x32 (4x2).
__global__ void gemm1_kernel(const _Float16* __restrict__ fgcl,
                             const _Float16* __restrict__ bgcl,
                             const float* __restrict__ norm_inv,
                             float* __restrict__ scores) {
    const int b = blockIdx.z;
    const int lane = threadIdx.x & 31, wv = threadIdx.x >> 5;
    const int m0 = blockIdx.y * 128 + (wv >> 2) * 64;
    const int n0 = blockIdx.x * 128 + (wv & 3) * 32;
    const int rA = lane & 15, kHalf = (lane >> 4) * 16;
    const _Float16* fgb = fgcl + (size_t)b * P2 * CC;
    const _Float16* bgb = bgcl + (size_t)b * P2 * CC;
    v8f acc[4][2] = {};
    for (int s = 0; s < 9; ++s) {
        const int dy = s / 3, dx = s % 3;
        const _Float16* ap[4];
        const _Float16* bp[2];
#pragma unroll
        for (int i = 0; i < 4; ++i) {
            const int mi = m0 + i * 16 + rA;
            ap[i] = fgb + (size_t)(((mi >> 5) + dy) * PW + (mi & 31) + dx) * CC + kHalf;
        }
#pragma unroll
        for (int j = 0; j < 2; ++j) {
            const int nj = n0 + j * 16 + rA;
            bp[j] = bgb + (size_t)(((nj >> 5) + dy) * PW + (nj & 31) + dx) * CC + kHalf;
        }
        __builtin_prefetch((const void*)ap[0], 0, 3);
        __builtin_prefetch((const void*)bp[0], 0, 3);
#pragma unroll
        for (int kc = 0; kc < 8; ++kc) {
            v16h a[4], bf[2];
#pragma unroll
            for (int i = 0; i < 4; ++i) a[i] = ldfrag(ap[i] + kc * 32);
#pragma unroll
            for (int j = 0; j < 2; ++j) bf[j] = ldfrag(bp[j] + kc * 32);
#pragma unroll
            for (int i = 0; i < 4; ++i)
#pragma unroll
                for (int j = 0; j < 2; ++j) acc[i][j] = wmma_f16(a[i], bf[j], acc[i][j]);
        }
    }
#pragma unroll
    for (int j = 0; j < 2; ++j) {
        const int n = n0 + j * 16 + rA;
        const float sc = norm_inv[b * HW + n];
#pragma unroll
        for (int i = 0; i < 4; ++i)
#pragma unroll
            for (int t = 0; t < 8; ++t) {
                const int m = m0 + i * 16 + t + ((lane >> 4) << 3);
                scores[((size_t)b * HW + m) * HW + n] = acc[i][j][t] * sc;
            }
    }
}

// --------------------- window-sum + softmax + norm-fold --------------------
// grid (1024, BZ), 256 thr: pooled = 3x3 spatial window sum of score rows,
// softmax over l, multiply by norm_inv[l], write f16 into padded attnp row.
__global__ void softmax_kernel(const float* __restrict__ scores,
                               const float* __restrict__ norm_inv,
                               _Float16* __restrict__ attnp) {
    const int b = blockIdx.y, m = blockIdx.x;
    const int h = m >> 5, w = m & 31;
    const int tid = threadIdx.x, wv = tid >> 5, ln = tid & 31;
    const float* sb = scores + (size_t)b * HW * HW;
    float pv[4] = {0.f, 0.f, 0.f, 0.f};
    for (int dy = -1; dy <= 1; ++dy)
        for (int dx = -1; dx <= 1; ++dx) {
            const int hh = h + dy, ww = w + dx;
            if (hh < 0 || hh >= HH || ww < 0 || ww >= HH) continue;
            const float* row = sb + (size_t)(hh * HH + ww) * HW;
#pragma unroll
            for (int q = 0; q < 4; ++q) pv[q] += row[tid + q * 256];
        }
    float mx = fmaxf(fmaxf(pv[0], pv[1]), fmaxf(pv[2], pv[3]));
#pragma unroll
    for (int o = 16; o > 0; o >>= 1) mx = fmaxf(mx, __shfl_xor(mx, o, 32));
    __shared__ float redm[8], reds[8];
    if (ln == 0) redm[wv] = mx;
    __syncthreads();
    mx = redm[0];
#pragma unroll
    for (int i = 1; i < 8; ++i) mx = fmaxf(mx, redm[i]);
    float sum = 0.f;
#pragma unroll
    for (int q = 0; q < 4; ++q) { pv[q] = __expf(pv[q] - mx); sum += pv[q]; }
#pragma unroll
    for (int o = 16; o > 0; o >>= 1) sum += __shfl_xor(sum, o, 32);
    if (ln == 0) reds[wv] = sum;
    __syncthreads();
    sum = 0.f;
#pragma unroll
    for (int i = 0; i < 8; ++i) sum += reds[i];
    const float inv = 1.f / sum;
    const float* ni = norm_inv + b * HW;
    _Float16* orow = attnp + ((size_t)b * P2 + (size_t)(h + 1) * PW + (w + 1)) * HW;
#pragma unroll
    for (int q = 0; q < 4; ++q) {
        const int l = tid + q * 256;
        orow[l] = (_Float16)(pv[q] * inv * ni[l]);
    }
}

// ------------------------------- GEMM2 -------------------------------------
// recovered[b][m][c] = sum_{i,j} sum_l attnp[(h+2-i,w+2-j)][l]*bgTs[j][c][ly+i][lx]
// epilogue: final = recovered*mask/9 + fg*(1-mask) -> padded f16 finalp
__global__ void gemm2_kernel(const _Float16* __restrict__ attnp,
                             const _Float16* __restrict__ bgTs,
                             const _Float16* __restrict__ fgcl,
                             const float* __restrict__ mask,
                             _Float16* __restrict__ finalp) {
    const int b = blockIdx.z;
    const int lane = threadIdx.x & 31, wv = threadIdx.x >> 5;
    const int m0 = blockIdx.y * 128 + (wv >> 2) * 64;
    const int n0 = blockIdx.x * 128 + (wv & 3) * 32;
    const int rA = lane & 15, kHalf = (lane >> 4) * 16;
    const _Float16* ab = attnp + (size_t)b * P2 * HW;
    v8f acc[4][2] = {};
    for (int s = 0; s < 9; ++s) {
        const int dy = s / 3, dx = s % 3;
        const _Float16* ap[4];
        const _Float16* bp[2];
#pragma unroll
        for (int i = 0; i < 4; ++i) {
            const int mi = m0 + i * 16 + rA;
            ap[i] = ab + (size_t)(((mi >> 5) + 2 - dy) * PW + (mi & 31) + 2 - dx) * HW + kHalf;
        }
#pragma unroll
        for (int j = 0; j < 2; ++j) {
            const int cn = n0 + j * 16 + rA;
            // base at ly=0: row (0+dy); kc adds 32 elems == one padded row
            bp[j] = bgTs + ((((size_t)dx * BZ + b) * CC + cn) * PW + dy) * HH + kHalf;
        }
        __builtin_prefetch((const void*)ap[0], 0, 3);
        for (int kc = 0; kc < 32; ++kc) {
            v16h a[4], bf[2];
#pragma unroll
            for (int i = 0; i < 4; ++i) a[i] = ldfrag(ap[i] + kc * 32);
#pragma unroll
            for (int j = 0; j < 2; ++j) bf[j] = ldfrag(bp[j] + (size_t)kc * 32);
#pragma unroll
            for (int i = 0; i < 4; ++i)
#pragma unroll
                for (int j = 0; j < 2; ++j) acc[i][j] = wmma_f16(a[i], bf[j], acc[i][j]);
        }
    }
    const float inv9 = 1.f / 9.f;
#pragma unroll
    for (int j = 0; j < 2; ++j) {
        const int cn = n0 + j * 16 + rA;
#pragma unroll
        for (int i = 0; i < 4; ++i)
#pragma unroll
            for (int t = 0; t < 8; ++t) {
                const int m = m0 + i * 16 + t + ((lane >> 4) << 3);
                const int h = m >> 5, w = m & 31;
                const float mk = mask[m];
                const float fgv =
                    (float)fgcl[((size_t)b * P2 + (size_t)(h + 1) * PW + (w + 1)) * CC + cn];
                const float val = acc[i][j][t] * mk * inv9 + fgv * (1.f - mk);
                finalp[((size_t)b * FP2 + (size_t)(h + 8) * FP + (w + 8)) * CC + cn] =
                    (_Float16)val;
            }
    }
}

// ------------------------------- GEMM3 -------------------------------------
// 4 dilated conv branches: out[b][g*64+oc][m] = relu(bias + sum_{s,c} ...)
// block 256 thr = 8 waves (4M x 2N); tile 128x64; wave 32x32 (2x2).
__global__ void gemm3_kernel(const _Float16* __restrict__ finalp,
                             const _Float16* __restrict__ wt,
                             const float* __restrict__ bias,
                             float* __restrict__ out) {
    const int z = blockIdx.z, b = z >> 2, g = z & 3;
    const int r = 1 << g;  // rates 1,2,4,8
    const int lane = threadIdx.x & 31, wv = threadIdx.x >> 5;
    const int m0 = blockIdx.y * 128 + (wv >> 1) * 32;
    const int n0 = (wv & 1) * 32;
    const int rA = lane & 15, kHalf = (lane >> 4) * 16;
    const _Float16* fb = finalp + (size_t)b * FP2 * CC;
    const _Float16* wb = wt + (size_t)g * 64 * KTOT;
    v8f acc[2][2] = {};
    for (int s = 0; s < 9; ++s) {
        const int dy = s / 3, dx = s % 3;
        const _Float16* ap[2];
        const _Float16* bp[2];
#pragma unroll
        for (int i = 0; i < 2; ++i) {
            const int mi = m0 + i * 16 + rA;
            const int arow = ((mi >> 5) + 8 + r * (dy - 1)) * FP + (mi & 31) + 8 + r * (dx - 1);
            ap[i] = fb + (size_t)arow * CC + kHalf;
        }
#pragma unroll
        for (int j = 0; j < 2; ++j) {
            const int oc = n0 + j * 16 + rA;
            bp[j] = wb + (size_t)oc * KTOT + s * CC + kHalf;
        }
        __builtin_prefetch((const void*)ap[0], 0, 3);
#pragma unroll
        for (int kc = 0; kc < 8; ++kc) {
            v16h a[2], bf[2];
#pragma unroll
            for (int i = 0; i < 2; ++i) a[i] = ldfrag(ap[i] + kc * 32);
#pragma unroll
            for (int j = 0; j < 2; ++j) bf[j] = ldfrag(bp[j] + kc * 32);
#pragma unroll
            for (int i = 0; i < 2; ++i)
#pragma unroll
                for (int j = 0; j < 2; ++j) acc[i][j] = wmma_f16(a[i], bf[j], acc[i][j]);
        }
    }
#pragma unroll
    for (int j = 0; j < 2; ++j) {
        const int oc = n0 + j * 16 + rA;
        const float bs = bias[g * 64 + oc];
#pragma unroll
        for (int i = 0; i < 2; ++i)
#pragma unroll
            for (int t = 0; t < 8; ++t) {
                const int m = m0 + i * 16 + t + ((lane >> 4) << 3);
                const float v = acc[i][j][t] + bs;
                out[((size_t)b * CC + g * 64 + oc) * HW + m] = v > 0.f ? v : 0.f;
            }
    }
}

// ------------------------------- launch ------------------------------------
extern "C" void kernel_launch(void* const* d_in, const int* in_sizes, int n_in,
                              void* d_out, int out_size, void* d_ws, size_t ws_size,
                              hipStream_t stream) {
    const float* fg   = (const float*)d_in[0];
    const float* mask = (const float*)d_in[1];
    const float* bg   = (const float*)d_in[2];
    const float* cw   = (const float*)d_in[3];
    const float* cb   = (const float*)d_in[4];
    float* out = (float*)d_out;

    char* ws = (char*)d_ws;
    _Float16* fgcl   = (_Float16*)(ws + OFF_FGCL);
    _Float16* bgcl   = (_Float16*)(ws + OFF_BGCL);
    _Float16* bgTs   = (_Float16*)(ws + OFF_BGTS);
    float*    ninv   = (float*)   (ws + OFF_NORM);
    float*    scores = (float*)   (ws + OFF_SCORE);
    _Float16* attnp  = (_Float16*)(ws + OFF_ATTNP);
    _Float16* finalp = (_Float16*)(ws + OFF_FINAL);
    _Float16* wt     = (_Float16*)(ws + OFF_WT);

    // zero the padded f16 staging buffers (borders must be 0 for WMMA frags)
    zero_kernel<<<4096, 256, 0, stream>>>((unsigned int*)attnp,
                                          (size_t)BZ * P2 * HW / 2);
    zero_kernel<<<4096, 256, 0, stream>>>((unsigned int*)finalp,
                                          (size_t)BZ * FP2 * CC / 2);

    prep_kernel<<<dim3(P2, BZ), 256, 0, stream>>>(fg, bg, mask, fgcl, bgcl, bgTs);
    wt_kernel<<<(4 * 64 * KTOT + 255) / 256, 256, 0, stream>>>(cw, wt);
    norm_kernel<<<dim3(HW, BZ), 256, 0, stream>>>(bgcl, ninv);

    gemm1_kernel<<<dim3(8, 8, BZ), 256, 0, stream>>>(fgcl, bgcl, ninv, scores);
    softmax_kernel<<<dim3(HW, BZ), 256, 0, stream>>>(scores, ninv, attnp);
    gemm2_kernel<<<dim3(2, 8, BZ), 256, 0, stream>>>(attnp, bgTs, fgcl, mask, finalp);
    gemm3_kernel<<<dim3(1, 8, BZ * 4), 256, 0, stream>>>(finalp, wt, cb, out);
}
